// static_hgt_56581899157999
// MI455X (gfx1250) — matrix operand order repaired and checked by reading, not compiled, not run on hardware
//
#include <hip/hip_runtime.h>
#include <hip/hip_bf16.h>
#include <math.h>

typedef __attribute__((ext_vector_type(16))) _Float16 v16h;
typedef __attribute__((ext_vector_type(8)))  float    v8f;

#define N_DOC   32000
#define N_TOPIC 800
#define N_WORD  80000
#define HDIM    128
#define NB      16

static inline int cdiv_i(long a, long b) { return (int)((a + b - 1) / b); }

// ---------------------------------------------------------------------------
// Gather-GEMM with WMMA f16 (f32 accumulate):
//   C[r, n] = act( alpha * sum_k A[gather(r), k] * W[k, n] + bias[n] )
// Templated on NT = N/16 (compile-time) so the N-tile loop fully unrolls.
// Block = 4 waves; wave w owns rows [16*(4*blk+w), +16) x ALL N columns.
//   - A fragment loaded straight from global (per ISA layout: lane {g,mr}
//     needs two contiguous 8-float runs at cols k0+8g and k0+16+8g) -> 4x b128.
//   - B panel (32 x N) staged once per block per K-step into LDS as f16 [n][k],
//     so each fragment gather is two contiguous 16-byte LDS reads.
// M must be a multiple of 16; K arbitrary (uniform-branch zero-pad tail).
// ---------------------------------------------------------------------------
template <int NT>
__global__ __launch_bounds__(128) void k_gemm_wmma(
    const float* __restrict__ A, int lda, const int* __restrict__ gather,
    const float* __restrict__ W, int ldw, const float* __restrict__ bias,
    float* __restrict__ C, int ldc,
    int M, int K, float alpha, int act)
{
    __shared__ _Float16 lb[NT * 16 * 32];          // B as [n][k], n = 0..16*NT-1
    const int lane = threadIdx.x & 31;
    const int wave = threadIdx.x >> 5;
    const int mtiles = M >> 4;
    const int mtile = blockIdx.x * 4 + wave;
    const bool active = mtile < mtiles;
    const int row0 = (active ? mtile : 0) * 16;
    const int g  = lane >> 4;                      // half-wave group
    const int mr = lane & 15;                      // row within tile / col within tile
    const int grow = row0 + mr;
    const int arow = gather ? gather[grow] : grow; // hoisted out of K loop
    const float* __restrict__ rp = A + (size_t)arow * lda;

    v8f acc[NT];
    #pragma unroll
    for (int i = 0; i < NT; ++i) acc[i] = (v8f){0.f,0.f,0.f,0.f,0.f,0.f,0.f,0.f};

    constexpr int NQ = NT * 4;                     // N/4

    for (int k0 = 0; k0 < K; k0 += 32) {
        const bool full = (k0 + 32) <= K;          // uniform per K-step
        // ---- cooperative B staging: NT float4 chunks per thread ----
        #pragma unroll
        for (int j = 0; j < NT; ++j) {
            const int idx4 = (int)threadIdx.x + 128 * j;
            const int n4 = (idx4 % NQ) * 4;
            const int kk = idx4 / NQ;
            float4 wv;
            if (full || (k0 + kk) < K)
                wv = *(const float4*)(W + (size_t)(k0 + kk) * ldw + n4);
            else
                wv = make_float4(0.f, 0.f, 0.f, 0.f);
            lb[(n4 + 0) * 32 + kk] = (_Float16)wv.x;
            lb[(n4 + 1) * 32 + kk] = (_Float16)wv.y;
            lb[(n4 + 2) * 32 + kk] = (_Float16)wv.z;
            lb[(n4 + 3) * 32 + kk] = (_Float16)wv.w;
        }
        __syncthreads();
        // ---- A fragment direct from global ----
        v16h av;
        const int c0 = k0 + g * 8;
        if (full) {
            const float4 a0 = *(const float4*)(rp + c0);
            const float4 a1 = *(const float4*)(rp + c0 + 4);
            const float4 a2 = *(const float4*)(rp + c0 + 16);
            const float4 a3 = *(const float4*)(rp + c0 + 20);
            av[0]=(_Float16)a0.x; av[1]=(_Float16)a0.y; av[2]=(_Float16)a0.z; av[3]=(_Float16)a0.w;
            av[4]=(_Float16)a1.x; av[5]=(_Float16)a1.y; av[6]=(_Float16)a1.z; av[7]=(_Float16)a1.w;
            av[8]=(_Float16)a2.x; av[9]=(_Float16)a2.y; av[10]=(_Float16)a2.z; av[11]=(_Float16)a2.w;
            av[12]=(_Float16)a3.x; av[13]=(_Float16)a3.y; av[14]=(_Float16)a3.z; av[15]=(_Float16)a3.w;
        } else {
            #pragma unroll
            for (int i = 0; i < 8; ++i)
                av[i] = (c0 + i < K) ? (_Float16)rp[c0 + i] : (_Float16)0.f;
            #pragma unroll
            for (int i = 0; i < 8; ++i)
                av[8 + i] = (c0 + 16 + i < K) ? (_Float16)rp[c0 + 16 + i] : (_Float16)0.f;
        }
        // ---- NT WMMAs sharing one A fragment ----
        #pragma unroll
        for (int nt = 0; nt < NT; ++nt) {
            v16h bv;
            #pragma unroll
            for (int i = 0; i < 16; ++i)
                bv[i] = lb[(nt * 16 + mr) * 32 + g * 8 + (i < 8 ? i : i + 8)];
            acc[nt] = __builtin_amdgcn_wmma_f32_16x16x32_f16(
                false, av, false, bv, (short)0, acc[nt], false, false);
        }
        __syncthreads();
    }

    if (active) {
        #pragma unroll
        for (int nt = 0; nt < NT; ++nt) {
            const int coln = nt * 16 + mr;
            const float bval = bias ? bias[coln] : 0.f;
            #pragma unroll
            for (int v = 0; v < 8; ++v) {
                float r = alpha * acc[nt][v] + bval;
                if (act == 1) r = tanhf(r);
                C[(size_t)(row0 + g * 8 + v) * ldc + coln] = r;
            }
        }
    }
}

// ---------------------------------------------------------------------------
// Small helper kernels
// ---------------------------------------------------------------------------
__global__ void k_fill(float* __restrict__ p, float v, long n)
{
    long i = (long)blockIdx.x * blockDim.x + threadIdx.x;
    if (i < n) p[i] = v;
}

__global__ void k_gather_rows(const float* __restrict__ src, const int* __restrict__ idx,
                              float* __restrict__ dst, int ncols, long n)
{
    long i = (long)blockIdx.x * blockDim.x + threadIdx.x;
    if (i < n) {
        const long r = i / ncols, c = i % ncols;
        dst[i] = src[(size_t)idx[r] * ncols + c];
    }
}

// pass 1: per-(edge,head) attention logit + segment max
__global__ void k_edge_att(const float* __restrict__ q, const float* __restrict__ ke,
                           const int* __restrict__ src, const int* __restrict__ dst,
                           const float* __restrict__ pri,
                           float* __restrict__ att_e, float* __restrict__ mbuf, int E)
{
    long t = (long)blockIdx.x * blockDim.x + threadIdx.x;
    if (t >= (long)E * 4) return;
    const int e = (int)(t >> 2), h = (int)(t & 3);
    const int s = src[e], d = dst[e];
    const float4* qp = (const float4*)(q  + (size_t)d * HDIM + h * 32);
    const float4* kp = (const float4*)(ke + (size_t)s * HDIM + h * 32);
    float acc = 0.f;
    #pragma unroll
    for (int j = 0; j < 8; ++j) {
        const float4 a = qp[j], b = kp[j];
        acc += a.x * b.x + a.y * b.y + a.z * b.z + a.w * b.w;
    }
    acc *= pri[h] * 0.17677669529663687f;   // 1/sqrt(32)
    att_e[(size_t)e * 4 + h] = acc;
    __hip_atomic_fetch_max(&mbuf[(size_t)d * 4 + h], acc,
                           __ATOMIC_RELAXED, __HIP_MEMORY_SCOPE_AGENT);
}

// pass 2: exp(att - max) + segment sum
__global__ void k_edge_exp(const float* __restrict__ mbuf, const int* __restrict__ dst,
                           float* __restrict__ att_e, float* __restrict__ sbuf, int E)
{
    long t = (long)blockIdx.x * blockDim.x + threadIdx.x;
    if (t >= (long)E * 4) return;
    const int e = (int)(t >> 2), h = (int)(t & 3);
    const int d = dst[e];
    const float ex = __expf(att_e[(size_t)e * 4 + h] - mbuf[(size_t)d * 4 + h]);
    att_e[(size_t)e * 4 + h] = ex;
    atomicAdd(&sbuf[(size_t)d * 4 + h], ex);
}

// pass 3: weighted message scatter
__global__ void k_edge_scat(const float* __restrict__ att_e, const float* __restrict__ sbuf,
                            const float* __restrict__ vm,
                            const int* __restrict__ src, const int* __restrict__ dst,
                            float* __restrict__ tout, int E)
{
    long t = (long)blockIdx.x * blockDim.x + threadIdx.x;
    if (t >= (long)E * 4) return;
    const int e = (int)(t >> 2), h = (int)(t & 3);
    const int s = src[e], d = dst[e];
    const float ssum = sbuf[(size_t)d * 4 + h];
    const float w = att_e[(size_t)e * 4 + h] / (ssum > 0.f ? ssum : 1.f);
    const float* vp = vm + (size_t)s * HDIM + h * 32;
    float* op = tout + (size_t)d * HDIM + h * 32;
    #pragma unroll 8
    for (int j = 0; j < 32; ++j) atomicAdd(&op[j], w * vp[j]);
}

// skip combine: h = tr * sigmoid(skip) + h * (1 - sigmoid(skip))
__global__ void k_comb(const float* __restrict__ tr, float* __restrict__ h,
                       const float* __restrict__ skip, int sidx, long n)
{
    long i = (long)blockIdx.x * blockDim.x + threadIdx.x;
    if (i < n) {
        const float a = 1.f / (1.f + __expf(-skip[sidx]));
        h[i] = tr[i] * a + h[i] * (1.f - a);
    }
}

__global__ void k_docmax(const float* __restrict__ hd, const int* __restrict__ gid,
                         float* __restrict__ g, long n)
{
    long i = (long)blockIdx.x * blockDim.x + threadIdx.x;
    if (i < n) {
        const long d = i / HDIM, j = i % HDIM;
        __hip_atomic_fetch_max(&g[(size_t)gid[d] * HDIM + j], hd[i],
                               __ATOMIC_RELAXED, __HIP_MEMORY_SCOPE_AGENT);
    }
}

__global__ void k_final(const float* __restrict__ g, const float* __restrict__ out_w,
                        const float* __restrict__ out_b, const float* __restrict__ y,
                        float* __restrict__ out)
{
    __shared__ float logits[NB];
    const int t = threadIdx.x;
    if (t < NB) {
        float acc = out_b[0];
        for (int j = 0; j < HDIM; ++j) acc += g[(size_t)t * HDIM + j] * out_w[j];
        logits[t] = acc;
    }
    __syncthreads();
    if (t == 0) {
        float loss = 0.f;
        for (int b = 0; b < NB; ++b) {
            const float l = logits[b], yb = y[b];
            // stable log_sigmoid(x) = min(x,0) - log1p(exp(-|x|))
            const float lsp = fminf(l, 0.f) - log1pf(__expf(-fabsf(l)));
            const float lsn = fminf(-l, 0.f) - log1pf(__expf(-fabsf(l)));
            loss += -(yb * lsp + (1.f - yb) * lsn);
            out[1 + b] = 1.f / (1.f + __expf(-l));
        }
        out[0] = loss / (float)NB;
    }
}

// ---------------------------------------------------------------------------
// Host-side driver
// ---------------------------------------------------------------------------
static void launch_gemm(hipStream_t s, const float* A, int lda, const int* gather,
                        const float* Wm, int ldw, const float* bias,
                        float* C, int ldc, int M, int N, int K, float alpha, int act)
{
    const int blocks = (M / 16 + 3) / 4;
    if (N == 128)
        k_gemm_wmma<8><<<blocks, 128, 0, s>>>(A, lda, gather, Wm, ldw, bias,
                                              C, ldc, M, K, alpha, act);
    else  // N == 32 (per-head relation transforms)
        k_gemm_wmma<2><<<blocks, 128, 0, s>>>(A, lda, gather, Wm, ldw, bias,
                                              C, ldc, M, K, alpha, act);
}

static void launch_fill(hipStream_t s, float* p, float v, long n)
{
    k_fill<<<cdiv_i(n, 256), 256, 0, s>>>(p, v, n);
}

struct RelDesc {
    const int* src; const int* dst; int E; int nsrc; int ndst;
    const float* ks; const float* vs; const float* qd; float* tout;
};

extern "C" void kernel_launch(void* const* d_in, const int* in_sizes, int n_in,
                              void* d_out, int out_size, void* d_ws, size_t ws_size,
                              hipStream_t stream)
{
    (void)in_sizes; (void)n_in; (void)out_size; (void)ws_size;

    const float* word_embeds  = (const float*)d_in[0];
    const float* topic_embeds = (const float*)d_in[1];
    const float* KW = (const float*)d_in[2];
    const float* Kb = (const float*)d_in[3];
    const float* QW = (const float*)d_in[4];
    const float* Qb = (const float*)d_in[5];
    const float* VW = (const float*)d_in[6];
    const float* Vb = (const float*)d_in[7];
    const float* AW = (const float*)d_in[8];
    const float* Ab = (const float*)d_in[9];
    const float* rel_att = (const float*)d_in[10];
    const float* rel_msg = (const float*)d_in[11];
    const float* rel_pri = (const float*)d_in[12];
    const float* skip    = (const float*)d_in[13];
    const float* adapt_w = (const float*)d_in[14];
    const float* adapt_b = (const float*)d_in[15];
    const float* out_w   = (const float*)d_in[16];
    const float* out_b   = (const float*)d_in[17];
    const float* y_data  = (const float*)d_in[18];
    const int* word_ids  = (const int*)d_in[19];
    const int* topic_ids = (const int*)d_in[20];
    const int* doc_gid   = (const int*)d_in[21];
    const int* td_src = (const int*)d_in[22]; const int* td_dst = (const int*)d_in[23];
    const int* tt_src = (const int*)d_in[24]; const int* tt_dst = (const int*)d_in[25];
    const int* wd_src = (const int*)d_in[26]; const int* wd_dst = (const int*)d_in[27];
    const int* wt_src = (const int*)d_in[28]; const int* wt_dst = (const int*)d_in[29];
    const int* ww_src = (const int*)d_in[30]; const int* ww_dst = (const int*)d_in[31];

    // Workspace bump allocator (float elements, 256B-aligned chunks)
    float* wsf = (float*)d_ws;
    size_t off = 0;
    auto alloc = [&](size_t n) -> float* {
        float* p = wsf + off;
        off += (n + 63) & ~(size_t)63;
        return p;
    };
    float* hw  = alloc((size_t)N_WORD * HDIM);
    float* ht  = alloc((size_t)N_TOPIC * HDIM);
    float* hd  = alloc((size_t)N_DOC * HDIM);
    float* q_d = alloc((size_t)N_DOC * HDIM);
    float* q_t = alloc((size_t)N_TOPIC * HDIM);
    float* q_w = alloc((size_t)N_WORD * HDIM);
    float* k_t = alloc((size_t)N_TOPIC * HDIM);
    float* v_t = alloc((size_t)N_TOPIC * HDIM);
    float* k_w = alloc((size_t)N_WORD * HDIM);
    float* v_w = alloc((size_t)N_WORD * HDIM);
    float* ke  = alloc((size_t)N_WORD * HDIM);       // per-relation transformed keys
    float* vm  = alloc((size_t)N_WORD * HDIM);       // per-relation transformed msgs
    float* t_d = alloc((size_t)N_DOC * HDIM);
    float* t_t = alloc((size_t)N_TOPIC * HDIM);
    float* t_w = alloc((size_t)N_WORD * HDIM);
    float* tmp = alloc((size_t)N_WORD * HDIM);       // combine GEMM output
    float* att_e = alloc((size_t)400000 * 4);
    float* mbuf  = alloc((size_t)N_WORD * 4);
    float* sbuf  = alloc((size_t)N_WORD * 4);
    float* gpool = alloc((size_t)NB * HDIM);

    // Node features
    launch_gemm(stream, word_embeds, 300, word_ids, adapt_w, HDIM, adapt_b,
                hw, HDIM, N_WORD, HDIM, 300, 1.f, /*tanh*/1);
    k_gather_rows<<<cdiv_i((long)N_TOPIC * HDIM, 256), 256, 0, stream>>>(
        topic_embeds, topic_ids, ht, HDIM, (long)N_TOPIC * HDIM);
    launch_fill(stream, hd, 0.f, (long)N_DOC * HDIM);

    for (int li = 0; li < 2; ++li) {
        auto Wat = [&](const float* base, int nid) { return base + ((size_t)li * 3 + nid) * HDIM * HDIM; };
        auto Bat = [&](const float* base, int nid) { return base + ((size_t)li * 3 + nid) * HDIM; };

        // Projections (Q for all types; K/V for topic & word sources)
        launch_gemm(stream, hd, HDIM, nullptr, Wat(QW, 0), HDIM, Bat(Qb, 0), q_d, HDIM, N_DOC,   HDIM, HDIM, 1.f, 0);
        launch_gemm(stream, ht, HDIM, nullptr, Wat(QW, 1), HDIM, Bat(Qb, 1), q_t, HDIM, N_TOPIC, HDIM, HDIM, 1.f, 0);
        launch_gemm(stream, hw, HDIM, nullptr, Wat(QW, 2), HDIM, Bat(Qb, 2), q_w, HDIM, N_WORD,  HDIM, HDIM, 1.f, 0);
        launch_gemm(stream, ht, HDIM, nullptr, Wat(KW, 1), HDIM, Bat(Kb, 1), k_t, HDIM, N_TOPIC, HDIM, HDIM, 1.f, 0);
        launch_gemm(stream, ht, HDIM, nullptr, Wat(VW, 1), HDIM, Bat(Vb, 1), v_t, HDIM, N_TOPIC, HDIM, HDIM, 1.f, 0);
        launch_gemm(stream, hw, HDIM, nullptr, Wat(KW, 2), HDIM, Bat(Kb, 2), k_w, HDIM, N_WORD,  HDIM, HDIM, 1.f, 0);
        launch_gemm(stream, hw, HDIM, nullptr, Wat(VW, 2), HDIM, Bat(Vb, 2), v_w, HDIM, N_WORD,  HDIM, HDIM, 1.f, 0);

        launch_fill(stream, t_d, 0.f, (long)N_DOC * HDIM);
        launch_fill(stream, t_t, 0.f, (long)N_TOPIC * HDIM);
        launch_fill(stream, t_w, 0.f, (long)N_WORD * HDIM);

        const RelDesc rels[5] = {
            { td_src, td_dst,  80000, N_TOPIC, N_DOC,   k_t, v_t, q_d, t_d },
            { tt_src, tt_dst,  20000, N_TOPIC, N_TOPIC, k_t, v_t, q_t, t_t },
            { wd_src, wd_dst, 250000, N_WORD,  N_DOC,   k_w, v_w, q_d, t_d },
            { wt_src, wt_dst, 100000, N_WORD,  N_TOPIC, k_w, v_w, q_t, t_t },
            { ww_src, ww_dst, 400000, N_WORD,  N_WORD,  k_w, v_w, q_w, t_w },
        };

        for (int r = 0; r < 5; ++r) {
            const RelDesc& R = rels[r];
            // Hoisted relation transforms: ke = k @ rel_att[li,r,h], vm = v @ rel_msg[li,r,h]
            for (int h = 0; h < 4; ++h) {
                const float* Ra = rel_att + (((size_t)li * 5 + r) * 4 + h) * 32 * 32;
                const float* Rm = rel_msg + (((size_t)li * 5 + r) * 4 + h) * 32 * 32;
                launch_gemm(stream, R.ks + h * 32, HDIM, nullptr, Ra, 32, nullptr,
                            ke + h * 32, HDIM, R.nsrc, 32, 32, 1.f, 0);
                launch_gemm(stream, R.vs + h * 32, HDIM, nullptr, Rm, 32, nullptr,
                            vm + h * 32, HDIM, R.nsrc, 32, 32, 1.f, 0);
            }
            launch_fill(stream, mbuf, -1e30f, (long)R.ndst * 4);
            launch_fill(stream, sbuf, 0.f,    (long)R.ndst * 4);
            const long nt = (long)R.E * 4;
            const float* pri = rel_pri + ((size_t)li * 5 + r) * 4;
            k_edge_att<<<cdiv_i(nt, 256), 256, 0, stream>>>(R.qd, ke, R.src, R.dst, pri, att_e, mbuf, R.E);
            k_edge_exp<<<cdiv_i(nt, 256), 256, 0, stream>>>(mbuf, R.dst, att_e, sbuf, R.E);
            k_edge_scat<<<cdiv_i(nt, 256), 256, 0, stream>>>(att_e, sbuf, vm, R.src, R.dst, R.tout, R.E);
        }

        // Combine with skip gate (alpha = 0.5 implements the cross-relation mean)
        struct { float* t; float* h; int n; float a; } combs[3] = {
            { t_d, hd, N_DOC,   0.5f },
            { t_t, ht, N_TOPIC, 0.5f },
            { t_w, hw, N_WORD,  1.0f },
        };
        for (int nid = 0; nid < 3; ++nid) {
            launch_gemm(stream, combs[nid].t, HDIM, nullptr, Wat(AW, nid), HDIM, Bat(Ab, nid),
                        tmp, HDIM, combs[nid].n, HDIM, HDIM, combs[nid].a, 0);
            const long n = (long)combs[nid].n * HDIM;
            k_comb<<<cdiv_i(n, 256), 256, 0, stream>>>(tmp, combs[nid].h, skip, li * 3 + nid, n);
        }
    }

    // Readout: per-group max over docs, linear head, BCE loss
    launch_fill(stream, gpool, -1e30f, (long)NB * HDIM);
    k_docmax<<<cdiv_i((long)N_DOC * HDIM, 256), 256, 0, stream>>>(hd, doc_gid, gpool, (long)N_DOC * HDIM);
    k_final<<<1, 64, 0, stream>>>(gpool, out_w, out_b, y_data, (float*)d_out);
}
